// WhaleAttention_60919816126690
// MI455X (gfx1250) — compile-verified, matrix operand order checked by reading, not run on hardware
//
#include <hip/hip_runtime.h>
#include <math.h>

// ---------------------------------------------------------------------------
// Types for CDNA5 WMMA (wave32)
// ---------------------------------------------------------------------------
typedef __attribute__((ext_vector_type(16))) __bf16        v16bf;
typedef __attribute__((ext_vector_type(8)))  float         v8f;
typedef __attribute__((ext_vector_type(4)))  unsigned int  u32x4;

union FragU {
    u32x4 q[2];
    v16bf bf;
};

// Per the CDNA5 16-bit A/B fragment layout (16x32 / 32x16):
//   lane l (l16 = l&15, half = l>>4) holds row r = base_row + l16,
//   K elements [half*8 .. half*8+7] and [16+half*8 .. 16+half*8+7].
// Both 8-element groups are contiguous bf16 => two 16-byte loads.
__device__ __forceinline__ v16bf load_frag(const unsigned short* p) {
    FragU f;
    const u32x4* q = (const u32x4*)p;
    f.q[0] = q[0];   // elements 0..7
    f.q[1] = q[2];   // elements 16..23 (offset +16 elems = +2 u32x4)
    return f.bf;
}

__device__ __forceinline__ v8f wmma_bf16(v16bf a, v16bf b, v8f c) {
    return __builtin_amdgcn_wmma_f32_16x16x32_bf16(
        /*neg_a=*/false, a, /*neg_b=*/false, b,
        /*c_mod=*/(short)0, c, /*reuse_a=*/false, /*reuse_b=*/false);
}

__device__ __forceinline__ unsigned short f2bf(float x) {
    unsigned int u = __float_as_uint(x);
    unsigned int r = (u + 0x7FFFu + ((u >> 16) & 1u)) >> 16;
    return (unsigned short)r;
}
__device__ __forceinline__ float bf2f(unsigned short b) {
    return __uint_as_float(((unsigned int)b) << 16);
}

// ---------------------------------------------------------------------------
// fp32 -> bf16 elementwise convert
// ---------------------------------------------------------------------------
__global__ __launch_bounds__(256) void cvt_bf16_kernel(const float* __restrict__ in,
                                                       unsigned short* __restrict__ out,
                                                       size_t n) {
    size_t i = (size_t)blockIdx.x * blockDim.x + threadIdx.x;
    if (i < n) out[i] = f2bf(in[i]);
}

// ---------------------------------------------------------------------------
// GEMM: out[M,Nw] = A[M,K](bf16) @ W[Nw,K]^T(bf16) + bias   (fp32 accumulate)
// Block: 256 threads = 8 waves; wave tile 16(M) x 64(N); block tile 64 x 128.
// ---------------------------------------------------------------------------
__global__ __launch_bounds__(256) void gemm_bf16_kernel(
    const unsigned short* __restrict__ A, const unsigned short* __restrict__ W,
    const float* __restrict__ bias, float* __restrict__ out,
    int M, int K, int Nw) {
    const int wave = threadIdx.x >> 5;
    const int lane = threadIdx.x & 31;
    const int half = lane >> 4;
    const int l16  = lane & 15;

    const int m0 = blockIdx.x * 64  + (wave >> 1) * 16;
    const int n0 = blockIdx.y * 128 + (wave & 1)  * 64;

    v8f acc0 = {0.f,0.f,0.f,0.f,0.f,0.f,0.f,0.f};
    v8f acc1 = acc0, acc2 = acc0, acc3 = acc0;

    const unsigned short* arow = A + (size_t)(m0 + l16) * K + half * 8;
    const unsigned short* wrow = W + (size_t)(n0 + l16) * K + half * 8;
    const size_t wstep = (size_t)16 * K;

    for (int k0 = 0; k0 < K; k0 += 32) {
        v16bf a  = load_frag(arow + k0);
        v16bf b0 = load_frag(wrow + k0);
        v16bf b1 = load_frag(wrow + wstep     + k0);
        v16bf b2 = load_frag(wrow + wstep * 2 + k0);
        v16bf b3 = load_frag(wrow + wstep * 3 + k0);
        acc0 = wmma_bf16(a, b0, acc0);
        acc1 = wmma_bf16(a, b1, acc1);
        acc2 = wmma_bf16(a, b2, acc2);
        acc3 = wmma_bf16(a, b3, acc3);
    }

    v8f accs[4] = {acc0, acc1, acc2, acc3};
#pragma unroll
    for (int t = 0; t < 4; ++t) {
        int col = n0 + t * 16 + l16;
        float bval = bias ? bias[col] : 0.0f;
#pragma unroll
        for (int j = 0; j < 8; ++j) {
            int row = m0 + half * 8 + j;
            out[(size_t)row * Nw + col] = accs[t][j] + bval;
        }
    }
}

// ---------------------------------------------------------------------------
// Row LayerNorm (fp32) -> bf16, optionally fused "+pos_bias_u/_v" dual output.
// One block (256 thr) per row of C=1024.
// ---------------------------------------------------------------------------
__global__ __launch_bounds__(256) void ln_kernel(
    const float* __restrict__ in, const float* __restrict__ g,
    const float* __restrict__ bb,
    const float* __restrict__ pbu, const float* __restrict__ pbv,
    unsigned short* __restrict__ out1, unsigned short* __restrict__ out2,
    int C) {
    const int row = blockIdx.x;
    const float* x = in + (size_t)row * C;

    float s = 0.f, s2 = 0.f;
    for (int c = threadIdx.x; c < C; c += blockDim.x) {
        float v = x[c];
        s += v; s2 += v * v;
    }
    for (int off = 16; off >= 1; off >>= 1) {
        s  += __shfl_xor(s,  off, 32);
        s2 += __shfl_xor(s2, off, 32);
    }
    __shared__ float red[16];
    __shared__ float mv[2];
    const int wave = threadIdx.x >> 5;
    const int lane = threadIdx.x & 31;
    if (lane == 0) { red[wave] = s; red[8 + wave] = s2; }
    __syncthreads();
    if (threadIdx.x == 0) {
        float ts = 0.f, ts2 = 0.f;
        for (int i = 0; i < 8; ++i) { ts += red[i]; ts2 += red[8 + i]; }
        float mean = ts / (float)C;
        float var  = ts2 / (float)C - mean * mean;
        mv[0] = mean;
        mv[1] = rsqrtf(var + 1e-6f);
    }
    __syncthreads();
    const float mean = mv[0], rstd = mv[1];
    for (int c = threadIdx.x; c < C; c += blockDim.x) {
        float y = (x[c] - mean) * rstd * g[c] + bb[c];
        size_t o = (size_t)row * C + c;
        if (out2) {
            out1[o] = f2bf(y + pbu[c]);   // pos_bias_u flat index == c (H*D == C)
            out2[o] = f2bf(y + pbv[c]);
        } else {
            out1[o] = f2bf(y);
        }
    }
}

// ---------------------------------------------------------------------------
// v[b,n,h*64+d] (fp32) -> Vt[b,h,d,n] (bf16)  so attn@V B-frags are contiguous
// ---------------------------------------------------------------------------
__global__ __launch_bounds__(256) void transpose_v_kernel(
    const float* __restrict__ v32, unsigned short* __restrict__ vt) {
    size_t idx = (size_t)blockIdx.x * blockDim.x + threadIdx.x;
    const size_t total = (size_t)4 * 2048 * 1024;
    if (idx >= total) return;
    int d = (int)(idx % 64);
    int h = (int)((idx / 64) % 16);
    size_t row = idx / 1024;          // b*N + n
    int n = (int)(row % 2048);
    int b = (int)(row / 2048);
    vt[(((size_t)b * 16 + h) * 64 + d) * 2048 + n] = f2bf(v32[idx]);
}

// ---------------------------------------------------------------------------
// Attention: per block = (b, h, 16-query tile).  4 waves / 128 threads.
// LDS: bf16 scores S[16][2048] (64 KB) + softmax stats.
// ---------------------------------------------------------------------------
__global__ __launch_bounds__(128) void attn_kernel(
    const unsigned short* __restrict__ qu, const unsigned short* __restrict__ qv,
    const unsigned short* __restrict__ kb, const unsigned short* __restrict__ pb,
    const unsigned short* __restrict__ vt, const unsigned char* __restrict__ mask,
    unsigned short* __restrict__ ao) {
    constexpr int Nn = 2048, C = 1024, H = 16, D = 64;
    constexpr float SCALE = 0.125f;   // D^-0.5

    extern __shared__ char smem[];
    unsigned short* Sb   = (unsigned short*)smem;              // [16][Nn]  64 KB
    float* redmax        = (float*)(smem + (size_t)16 * Nn * 2); // [4][16]
    float* rmax          = redmax + 64;                        // [16]
    float* redsum        = rmax + 16;                          // [16][8]
    float* rsum          = redsum + 128;                       // [16]

    const int nt = blockIdx.x, h = blockIdx.y, b = blockIdx.z;
    const int wave = threadIdx.x >> 5;
    const int lane = threadIdx.x & 31;
    const int half = lane >> 4;
    const int l16  = lane & 15;
    const int n0   = nt * 16;

    const size_t basec = ((size_t)b * Nn) * C + (size_t)h * D;

    // ---- A fragments for Q_u and Q_v (D=64 => two K=32 chunks each) ----
    const unsigned short* qurow = qu + basec + (size_t)(n0 + l16) * C + half * 8;
    const unsigned short* qvrow = qv + basec + (size_t)(n0 + l16) * C + half * 8;
    v16bf au0 = load_frag(qurow),      au1 = load_frag(qurow + 32);
    v16bf av0 = load_frag(qvrow),      av1 = load_frag(qvrow + 32);

    // ---- Pass A: score tiles -> LDS (bf16), track row max ----
    float pmax[8];
#pragma unroll
    for (int j = 0; j < 8; ++j) pmax[j] = -3.0e38f;

    for (int mt = wave; mt < Nn / 16; mt += 4) {
        const int m0 = mt * 16;
        const unsigned short* krow = kb + basec + (size_t)(m0 + l16) * C + half * 8;
        const unsigned short* prow = pb + basec + (size_t)(m0 + l16) * C + half * 8;
        v16bf bk0 = load_frag(krow), bk1 = load_frag(krow + 32);
        v16bf bp0 = load_frag(prow), bp1 = load_frag(prow + 32);
        v8f acc = {0.f,0.f,0.f,0.f,0.f,0.f,0.f,0.f};
        acc = wmma_bf16(au0, bk0, acc);
        acc = wmma_bf16(au1, bk1, acc);
        acc = wmma_bf16(av0, bp0, acc);
        acc = wmma_bf16(av1, bp1, acc);
#pragma unroll
        for (int j = 0; j < 8; ++j) {
            float sv = acc[j] * SCALE;
            pmax[j] = fmaxf(pmax[j], sv);
            Sb[(size_t)(half * 8 + j) * Nn + m0 + l16] = f2bf(sv);
        }
    }
    // reduce max across the 16 lanes of each half-wave
#pragma unroll
    for (int j = 0; j < 8; ++j)
        for (int off = 8; off >= 1; off >>= 1)
            pmax[j] = fmaxf(pmax[j], __shfl_xor(pmax[j], off, 32));
    if (l16 == 0) {
#pragma unroll
        for (int j = 0; j < 8; ++j) redmax[wave * 16 + half * 8 + j] = pmax[j];
    }
    __syncthreads();
    if (threadIdx.x < 16) {
        int r = threadIdx.x;
        float m = fmaxf(fmaxf(redmax[r], redmax[16 + r]),
                        fmaxf(redmax[32 + r], redmax[48 + r]));
        rmax[r] = m;
    }
    __syncthreads();

    // ---- Pass B: exp / mask / row-sum (in-place bf16) ----
    {
        const int r = threadIdx.x >> 3;
        const int seg = threadIdx.x & 7;
        const float rm = rmax[r];
        const size_t mbase = (size_t)b * Nn;
        float ssum = 0.f;
        const int m1 = (seg + 1) * (Nn / 8);
        for (int m = seg * (Nn / 8); m < m1; ++m) {
            float sv = bf2f(Sb[(size_t)r * Nn + m]);
            float e = __expf(sv - rm);
            e = mask[mbase + m] ? e : 0.f;
            Sb[(size_t)r * Nn + m] = f2bf(e);
            ssum += e;
        }
        redsum[threadIdx.x] = ssum;
    }
    __syncthreads();
    if (threadIdx.x < 16) {
        float t = 0.f;
        for (int i = 0; i < 8; ++i) t += redsum[threadIdx.x * 8 + i];
        rsum[threadIdx.x] = t;
    }
    __syncthreads();

    // ---- Pass C: O = P @ V^T; each wave owns one 16-wide d tile ----
    const int d0 = wave * 16;
    v8f o = {0.f,0.f,0.f,0.f,0.f,0.f,0.f,0.f};
    const unsigned short* vrow = vt + ((size_t)(b * H + h) * D + d0 + l16) * Nn + half * 8;
    const unsigned short* srow = Sb + (size_t)l16 * Nn + half * 8;
    for (int k0 = 0; k0 < Nn; k0 += 32) {
        v16bf a  = load_frag(srow + k0);   // LDS -> ds_load_b128 pairs
        v16bf bv = load_frag(vrow + k0);   // global, contiguous along n
        o = wmma_bf16(a, bv, o);
    }
#pragma unroll
    for (int j = 0; j < 8; ++j) {
        const int r = half * 8 + j;
        const float val = o[j] / rsum[r];
        const int n = n0 + r;
        ao[((size_t)b * Nn + n) * C + (size_t)h * D + d0 + l16] = f2bf(val);
    }
}

// ---------------------------------------------------------------------------
// Host-side launcher
// ---------------------------------------------------------------------------
extern "C" void kernel_launch(void* const* d_in, const int* in_sizes, int n_in,
                              void* d_out, int out_size, void* d_ws, size_t ws_size,
                              hipStream_t stream) {
    (void)in_sizes; (void)n_in; (void)out_size; (void)ws_size;

    const float* x    = (const float*)d_in[0];
    const unsigned char* mask = (const unsigned char*)d_in[1];
    const float* pos  = (const float*)d_in[2];
    const float* Wq   = (const float*)d_in[3];
    const float* bq   = (const float*)d_in[4];
    const float* Wk   = (const float*)d_in[5];
    const float* bk   = (const float*)d_in[6];
    const float* Wv   = (const float*)d_in[7];
    const float* bv   = (const float*)d_in[8];
    const float* Wp   = (const float*)d_in[9];
    const float* pbu  = (const float*)d_in[10];
    const float* pbv  = (const float*)d_in[11];
    const float* gq   = (const float*)d_in[12];
    const float* bqln = (const float*)d_in[13];
    const float* gk   = (const float*)d_in[14];
    const float* bkln = (const float*)d_in[15];
    const float* Wo   = (const float*)d_in[16];
    const float* bo   = (const float*)d_in[17];
    float* out        = (float*)d_out;

    constexpr int Bb = 4, Nn = 2048, C = 1024, H = 16;
    constexpr size_t MN = (size_t)Bb * Nn;       // 8192 rows
    const size_t actElems = MN * C;              // 8,388,608
    const size_t wElems   = (size_t)C * C;       // 1,048,576

    char* ws = (char*)d_ws;
    size_t off = 0;
    auto alloc = [&](size_t bytes) -> char* {
        char* p = ws + off;
        off += (bytes + 255) & ~(size_t)255;
        return p;
    };

    unsigned short* x_bf   = (unsigned short*)alloc(actElems * 2);
    unsigned short* pos_bf = (unsigned short*)alloc(actElems * 2);
    unsigned short* wq_bf  = (unsigned short*)alloc(wElems * 2);
    unsigned short* wk_bf  = (unsigned short*)alloc(wElems * 2);
    unsigned short* wv_bf  = (unsigned short*)alloc(wElems * 2);
    unsigned short* wp_bf  = (unsigned short*)alloc(wElems * 2);
    unsigned short* wo_bf  = (unsigned short*)alloc(wElems * 2);
    float* q32 = (float*)alloc(actElems * 4);
    float* k32 = (float*)alloc(actElems * 4);
    float* v32 = (float*)alloc(actElems * 4);
    float* p32 = (float*)alloc(actElems * 4);
    unsigned short* qu_bf = (unsigned short*)alloc(actElems * 2);
    unsigned short* qv_bf = (unsigned short*)alloc(actElems * 2);
    unsigned short* k_bf  = (unsigned short*)alloc(actElems * 2);
    // Aliases (lifetimes proven disjoint by kernel ordering below):
    unsigned short* p_bf  = (unsigned short*)q32;   // written after LN(q) consumes q32
    unsigned short* vt_bf = pos_bf;                 // pos_bf dead after p GEMM
    unsigned short* ao_bf = x_bf;                   // x_bf dead after q/k/v GEMMs

    const int cvtBlocks = (int)((actElems + 255) / 256);
    const int wCvtBlocks = (int)((wElems + 255) / 256);

    // Stage 0: fp32 -> bf16 conversions
    cvt_bf16_kernel<<<cvtBlocks, 256, 0, stream>>>(x,   x_bf,   actElems);
    cvt_bf16_kernel<<<cvtBlocks, 256, 0, stream>>>(pos, pos_bf, actElems);
    cvt_bf16_kernel<<<wCvtBlocks, 256, 0, stream>>>(Wq, wq_bf, wElems);
    cvt_bf16_kernel<<<wCvtBlocks, 256, 0, stream>>>(Wk, wk_bf, wElems);
    cvt_bf16_kernel<<<wCvtBlocks, 256, 0, stream>>>(Wv, wv_bf, wElems);
    cvt_bf16_kernel<<<wCvtBlocks, 256, 0, stream>>>(Wp, wp_bf, wElems);
    cvt_bf16_kernel<<<wCvtBlocks, 256, 0, stream>>>(Wo, wo_bf, wElems);

    // Stage 1: projections (WMMA bf16, fp32 accumulate)
    dim3 ggrid((unsigned)(MN / 64), (unsigned)(C / 128));
    gemm_bf16_kernel<<<ggrid, 256, 0, stream>>>(x_bf,   wq_bf, bq,      q32, (int)MN, C, C);
    gemm_bf16_kernel<<<ggrid, 256, 0, stream>>>(x_bf,   wk_bf, bk,      k32, (int)MN, C, C);
    gemm_bf16_kernel<<<ggrid, 256, 0, stream>>>(x_bf,   wv_bf, bv,      v32, (int)MN, C, C);
    gemm_bf16_kernel<<<ggrid, 256, 0, stream>>>(pos_bf, wp_bf, nullptr, p32, (int)MN, C, C);

    // Stage 2: LayerNorm + pos-bias fusion; V transpose; P convert
    ln_kernel<<<(unsigned)MN, 256, 0, stream>>>(q32, gq, bqln, pbu, pbv, qu_bf, qv_bf, C);
    ln_kernel<<<(unsigned)MN, 256, 0, stream>>>(k32, gk, bkln, nullptr, nullptr, k_bf, nullptr, C);
    cvt_bf16_kernel<<<cvtBlocks, 256, 0, stream>>>(p32, p_bf, actElems);      // into old q32
    transpose_v_kernel<<<cvtBlocks, 256, 0, stream>>>(v32, vt_bf);            // into old pos_bf

    // Stage 3: attention (scores + softmax + O), all contractions via WMMA
    {
        dim3 agrid((unsigned)(Nn / 16), (unsigned)H, (unsigned)Bb);
        size_t shmem = (size_t)16 * Nn * 2 + (64 + 16 + 128 + 16) * sizeof(float);
        attn_kernel<<<agrid, 128, shmem, stream>>>(qu_bf, qv_bf, k_bf, p_bf, vt_bf, mask, ao_bf);
    }

    // Stage 4: output projection -> fp32 d_out
    gemm_bf16_kernel<<<ggrid, 256, 0, stream>>>(ao_bf, wo_bf, bo, out, (int)MN, C, C);
}
